// m_block_p_1w1a_3719441678864
// MI455X (gfx1250) — compile-verified, matrix operand order checked by reading, not run on hardware
//
#include <hip/hip_runtime.h>

typedef __attribute__((ext_vector_type(8))) int v8i;

#define EPSV 1e-5f
#define B_   128
#define CIN  64
#define H_   56
#define W_   56
#define HW_  (H_ * W_)                 // 3136
#define NPOS ((size_t)B_ * HW_)        // 401408
#define XB_BYTES ((unsigned long long)NPOS * 64ull)   // 25,690,112
#define WP_OFF   (2ull * XB_BYTES)
#define BNP_OFF  (WP_OFF + 22528ull)

// packed-weight sub-offsets (bytes inside Wp)
#define WP_STEM 0       // 2 tiles  (32 out-ch, K=64)
#define WP_B1   2048    // 9 tiles  [tap*3 + mtile] (48 out-ch, K=64 padded)
#define WP_B2   11264   // 9 tiles
#define WP_B3   20480   // 2 tiles
// bn params (floats inside bnp): inv1[32] b1[32] inv2[48] b2[48] inv3[48] b3[48] inv4[32] b4[32]

__device__ __forceinline__ v8i wmma_iu8(v8i a, v8i b, v8i c) {
    return __builtin_amdgcn_wmma_i32_16x16x64_iu8(true, a, true, b, c, false, false);
}

// ---------------------------------------------------------------------------
// Kernel 1: binarize x (fp32 NCHW) -> Xb int8 [b,h,w,c] (+1/-1/0)
// Block: 256 threads handles one (b, 64-position chunk). 3136 = 49*64 exact.
// ---------------------------------------------------------------------------
__global__ void __launch_bounds__(256) binarize_x_kernel(const float* __restrict__ x,
                                                         signed char* __restrict__ Xb) {
    __shared__ signed char tile[64 * 64];
    int blk = blockIdx.x;
    int b = blk / 49;
    int hw0 = (blk % 49) * 64;
    int t = threadIdx.x;
    int c = t >> 2;          // 0..63
    int j = t & 3;           // 0..3
    const float* src = x + ((size_t)b * CIN + c) * HW_ + hw0 + j * 16;
#pragma unroll
    for (int k = 0; k < 16; ++k) {
        float v = src[k];
        tile[(j * 16 + k) * 64 + c] = (signed char)((v > 0.f) - (v < 0.f));
    }
    __syncthreads();
    int pos = t >> 2;        // 0..63
    int q = t & 3;           // 16-byte quarter
    *(uint4*)(Xb + ((size_t)b * HW_ + hw0 + pos) * 64 + q * 16) =
        *(const uint4*)(tile + pos * 64 + q * 16);
}

// ---------------------------------------------------------------------------
// Kernel 2: pack/binarize weights into WMMA tile blobs + precompute BN params
// blob[lane*32 + j] = signrow[m = lane&15][(lane>>4)*32 + j]
// ---------------------------------------------------------------------------
__global__ void __launch_bounds__(256) pack_w_kernel(
    const float* __restrict__ w1, const float* __restrict__ w2,
    const float* __restrict__ w3, const float* __restrict__ w4,
    const float* __restrict__ g1, const float* __restrict__ bb1,
    const float* __restrict__ m1, const float* __restrict__ v1,
    const float* __restrict__ g2, const float* __restrict__ bb2,
    const float* __restrict__ m2, const float* __restrict__ v2,
    const float* __restrict__ g3, const float* __restrict__ bb3,
    const float* __restrict__ m3, const float* __restrict__ v3,
    const float* __restrict__ g4, const float* __restrict__ bb4,
    const float* __restrict__ m4, const float* __restrict__ v4,
    signed char* __restrict__ wp, float* __restrict__ bnp) {
    int tid = threadIdx.x;

    // stem: 2 tiles, K = 64 real channels
    for (int t = tid; t < 2048; t += 256) {
        int tileid = t / 1024, i = t % 1024;
        int lane = i / 32, j = i % 32;
        int oc = tileid * 16 + (lane & 15);
        int ch = (lane >> 4) * 32 + j;
        float wv = w1[oc * 64 + ch];
        wp[WP_STEM + t] = (signed char)((wv > 0.f) - (wv < 0.f));
    }
    // branch1: w2 [48,32,3,1] -> 9 tiles (tap*3+mt), K padded 32->64
    for (int t = tid; t < 9216; t += 256) {
        int blob = t / 1024, i = t % 1024;
        int tap = blob / 3, mt = blob % 3;
        int lane = i / 32, j = i % 32;
        int oc = mt * 16 + (lane & 15);
        int ch = (lane >> 4) * 32 + j;
        signed char s = 0;
        if (ch < 32) {
            float wv = w2[(oc * 32 + ch) * 3 + tap];
            s = (signed char)((wv > 0.f) - (wv < 0.f));
        }
        wp[WP_B1 + t] = s;
    }
    // branch2: w3 [48,32,1,3]
    for (int t = tid; t < 9216; t += 256) {
        int blob = t / 1024, i = t % 1024;
        int tap = blob / 3, mt = blob % 3;
        int lane = i / 32, j = i % 32;
        int oc = mt * 16 + (lane & 15);
        int ch = (lane >> 4) * 32 + j;
        signed char s = 0;
        if (ch < 32) {
            float wv = w3[(oc * 32 + ch) * 3 + tap];
            s = (signed char)((wv > 0.f) - (wv < 0.f));
        }
        wp[WP_B2 + t] = s;
    }
    // branch3: w4 [32,32,1,1] -> 2 tiles
    for (int t = tid; t < 2048; t += 256) {
        int tileid = t / 1024, i = t % 1024;
        int lane = i / 32, j = i % 32;
        int oc = tileid * 16 + (lane & 15);
        int ch = (lane >> 4) * 32 + j;
        signed char s = 0;
        if (ch < 32) {
            float wv = w4[oc * 32 + ch];
            s = (signed char)((wv > 0.f) - (wv < 0.f));
        }
        wp[WP_B3 + t] = s;
    }
    // BN fold: inv = g/sqrt(v+eps), beta = b - m*inv
    if (tid < 32) {
        float inv = g1[tid] / sqrtf(v1[tid] + EPSV);
        bnp[tid] = inv;  bnp[32 + tid] = bb1[tid] - m1[tid] * inv;
        float inv4 = g4[tid] / sqrtf(v4[tid] + EPSV);
        bnp[256 + tid] = inv4; bnp[288 + tid] = bb4[tid] - m4[tid] * inv4;
    }
    if (tid < 48) {
        float inv2 = g2[tid] / sqrtf(v2[tid] + EPSV);
        bnp[64 + tid] = inv2;  bnp[112 + tid] = bb2[tid] - m2[tid] * inv2;
        float inv3 = g3[tid] / sqrtf(v3[tid] + EPSV);
        bnp[160 + tid] = inv3; bnp[208 + tid] = bb3[tid] - m3[tid] * inv3;
    }
}

// ---------------------------------------------------------------------------
// Kernel 3: stem 1x1 conv (K=64) + BN + ReLU + binarize -> Hb {0,1} int8 rows
// 1 wave per 16 positions, 2 WMMAs (32 out channels). 3136 blocks x 8 waves.
// ---------------------------------------------------------------------------
__global__ void __launch_bounds__(256) stem_kernel(const signed char* __restrict__ Xb,
                                                   signed char* __restrict__ Hb,
                                                   const signed char* __restrict__ wp,
                                                   const float* __restrict__ bnp) {
    int wave = threadIdx.x >> 5, lane = threadIdx.x & 31;
    size_t tileid = (size_t)blockIdx.x * 8 + wave;
    if (tileid * 16 >= NPOS) return;           // wave-uniform
    size_t p0 = tileid * 16;
    int n = lane & 15, half = lane >> 4;

    v8i bf = *(const v8i*)(Xb + (p0 + n) * 64 + half * 32);
    v8i a0 = *(const v8i*)(wp + WP_STEM + 0 * 1024 + lane * 32);
    v8i a1 = *(const v8i*)(wp + WP_STEM + 1 * 1024 + lane * 32);
    v8i z = {0, 0, 0, 0, 0, 0, 0, 0};
    v8i c0 = wmma_iu8(a0, bf, z);
    v8i c1 = wmma_iu8(a1, bf, z);

    const float* inv = bnp;
    const float* beta = bnp + 32;
    unsigned long long p0bits = 0ull, p1bits = 0ull;
#pragma unroll
    for (int r = 0; r < 8; ++r) {
        int ch = r + 8 * half;
        float f0 = (float)c0[r] * inv[ch] + beta[ch];
        float f1 = (float)c1[r] * inv[ch + 16] + beta[ch + 16];
        p0bits |= (unsigned long long)(f0 > 0.f ? 1 : 0) << (8 * r);
        p1bits |= (unsigned long long)(f1 > 0.f ? 1 : 0) << (8 * r);
    }
    signed char* row = Hb + (p0 + n) * 64;
    *(unsigned long long*)(row + 8 * half) = p0bits;        // ch 0-7 / 8-15
    *(unsigned long long*)(row + 16 + 8 * half) = p1bits;   // ch 16-23 / 24-31
    *(unsigned long long*)(row + 32 + 8 * half) = 0ull;     // K pad
    *(unsigned long long*)(row + 48 + 8 * half) = 0ull;
}

// ---------------------------------------------------------------------------
// Kernel 4: branch1 (3,1) conv + BN + ReLU + maxpool(1,3)/(1,2) -> out ch 0-47
// 1 block per (b,h) row; 4 waves cover w-tiles; pool via LDS.
// ---------------------------------------------------------------------------
__global__ void __launch_bounds__(128) branch1_kernel(const signed char* __restrict__ Hb,
                                                      const signed char* __restrict__ wp,
                                                      const float* __restrict__ bnp,
                                                      float* __restrict__ out) {
    __shared__ float smem[48 * 64];
    int rowid = blockIdx.x;
    int b = rowid / H_, h = rowid % H_;
    int wave = threadIdx.x >> 5, lane = threadIdx.x & 31;
    int n = lane & 15, half = lane >> 4;
    int wx = wave * 16 + n;                    // input w position 0..63 (valid < 56)

    v8i z = {0, 0, 0, 0, 0, 0, 0, 0};
    v8i acc[3] = {z, z, z};
#pragma unroll
    for (int tap = 0; tap < 3; ++tap) {
        int hh = h + tap - 1;
        v8i bf = z;
        if (hh >= 0 && hh < H_ && wx < W_)
            bf = *(const v8i*)(Hb + (((size_t)b * H_ + hh) * W_ + wx) * 64 + half * 32);
#pragma unroll
        for (int mt = 0; mt < 3; ++mt) {
            v8i a = *(const v8i*)(wp + WP_B1 + (tap * 3 + mt) * 1024 + lane * 32);
            acc[mt] = wmma_iu8(a, bf, acc[mt]);
        }
    }
    const float* inv = bnp + 64;
    const float* beta = bnp + 112;
    if (wx < W_) {
#pragma unroll
        for (int mt = 0; mt < 3; ++mt)
#pragma unroll
            for (int r = 0; r < 8; ++r) {
                int c = mt * 16 + r + 8 * half;
                float f = (float)acc[mt][r] * inv[c] + beta[c];
                smem[c * 64 + wx] = fmaxf(f, 0.f);
            }
    }
    __syncthreads();
    // maxpool (1,3) stride (1,2) pad (0,1); post-ReLU values >= 0 so pad=0 is safe
    for (int i = threadIdx.x; i < 48 * 28; i += blockDim.x) {
        int c = i / 28, wq = i % 28;
        int wi = 2 * wq;
        float m = smem[c * 64 + wi];
        if (wi > 0) m = fmaxf(m, smem[c * 64 + wi - 1]);
        m = fmaxf(m, smem[c * 64 + wi + 1]);   // wi+1 <= 55
        out[(((size_t)b * 128 + c) * H_ + h) * 28 + wq] = m;
    }
}

// ---------------------------------------------------------------------------
// Kernel 5: branch2 (1,3)/s(1,2) -> ch 48-95  and branch3 1x1/s(1,2) -> ch 96-127
// Shared strided B-fragments; 1 block (2 waves) per (b,h) row.
// ---------------------------------------------------------------------------
__global__ void __launch_bounds__(64) branch23_kernel(const signed char* __restrict__ Hb,
                                                      const signed char* __restrict__ wp,
                                                      const float* __restrict__ bnp,
                                                      float* __restrict__ out) {
    int rowid = blockIdx.x;
    int b = rowid / H_, h = rowid % H_;
    int wave = threadIdx.x >> 5, lane = threadIdx.x & 31;
    int n = lane & 15, half = lane >> 4;
    int wq = wave * 16 + n;                    // output w' 0..31 (valid < 28)
    size_t rowbase = ((size_t)b * H_ + h) * W_;

    v8i z = {0, 0, 0, 0, 0, 0, 0, 0};
    v8i bf[3];
#pragma unroll
    for (int tap = 0; tap < 3; ++tap) {
        int w = 2 * wq + tap - 1;
        v8i t = z;
        if (wq < 28 && w >= 0 && w < W_)
            t = *(const v8i*)(Hb + (rowbase + w) * 64 + half * 32);
        bf[tap] = t;
    }
    v8i acc2[3] = {z, z, z};
#pragma unroll
    for (int tap = 0; tap < 3; ++tap)
#pragma unroll
        for (int mt = 0; mt < 3; ++mt) {
            v8i a = *(const v8i*)(wp + WP_B2 + (tap * 3 + mt) * 1024 + lane * 32);
            acc2[mt] = wmma_iu8(a, bf[tap], acc2[mt]);
        }
    v8i acc3[2];
#pragma unroll
    for (int mt = 0; mt < 2; ++mt) {
        v8i a = *(const v8i*)(wp + WP_B3 + mt * 1024 + lane * 32);
        acc3[mt] = wmma_iu8(a, bf[1], z);      // center tap: w = 2*w'
    }
    if (wq < 28) {
        const float* inv3 = bnp + 160; const float* beta3 = bnp + 208;
        const float* inv4 = bnp + 256; const float* beta4 = bnp + 288;
#pragma unroll
        for (int mt = 0; mt < 3; ++mt)
#pragma unroll
            for (int r = 0; r < 8; ++r) {
                int c = mt * 16 + r + 8 * half;
                float f = fmaxf((float)acc2[mt][r] * inv3[c] + beta3[c], 0.f);
                out[(((size_t)b * 128 + 48 + c) * H_ + h) * 28 + wq] = f;
            }
#pragma unroll
        for (int mt = 0; mt < 2; ++mt)
#pragma unroll
            for (int r = 0; r < 8; ++r) {
                int c = mt * 16 + r + 8 * half;
                float f = fmaxf((float)acc3[mt][r] * inv4[c] + beta4[c], 0.f);
                out[(((size_t)b * 128 + 96 + c) * H_ + h) * 28 + wq] = f;
            }
    }
}

// ---------------------------------------------------------------------------
extern "C" void kernel_launch(void* const* d_in, const int* in_sizes, int n_in,
                              void* d_out, int out_size, void* d_ws, size_t ws_size,
                              hipStream_t stream) {
    const float* x  = (const float*)d_in[0];
    const float* w1 = (const float*)d_in[1];
    const float* w2 = (const float*)d_in[2];
    const float* w3 = (const float*)d_in[3];
    const float* w4 = (const float*)d_in[4];
    const float* g1 = (const float*)d_in[5],  *bb1 = (const float*)d_in[6];
    const float* m1 = (const float*)d_in[7],  *v1  = (const float*)d_in[8];
    const float* g2 = (const float*)d_in[9],  *bb2 = (const float*)d_in[10];
    const float* m2 = (const float*)d_in[11], *v2  = (const float*)d_in[12];
    const float* g3 = (const float*)d_in[13], *bb3 = (const float*)d_in[14];
    const float* m3 = (const float*)d_in[15], *v3  = (const float*)d_in[16];
    const float* g4 = (const float*)d_in[17], *bb4 = (const float*)d_in[18];
    const float* m4 = (const float*)d_in[19], *v4  = (const float*)d_in[20];
    float* out = (float*)d_out;

    char* ws = (char*)d_ws;
    signed char* Xb = (signed char*)ws;
    signed char* Hb = (signed char*)(ws + XB_BYTES);
    signed char* Wp = (signed char*)(ws + WP_OFF);
    float* bnp = (float*)(ws + BNP_OFF);

    binarize_x_kernel<<<B_ * 49, 256, 0, stream>>>(x, Xb);
    pack_w_kernel<<<1, 256, 0, stream>>>(w1, w2, w3, w4,
                                         g1, bb1, m1, v1, g2, bb2, m2, v2,
                                         g3, bb3, m3, v3, g4, bb4, m4, v4,
                                         Wp, bnp);
    stem_kernel<<<3136, 256, 0, stream>>>(Xb, Hb, Wp, bnp);
    branch1_kernel<<<B_ * H_, 128, 0, stream>>>(Hb, Wp, bnp, out);
    branch23_kernel<<<B_ * H_, 64, 0, stream>>>(Hb, Wp, bnp, out);
}
